// LightGT_40544491274544
// MI455X (gfx1250) — compile-verified, matrix operand order checked by reading, not run on hardware
//
#include <hip/hip_runtime.h>
#include <hip/hip_bf16.h>

typedef __attribute__((ext_vector_type(16))) _Float16 v16h;
typedef __attribute__((ext_vector_type(8)))  _Float16 v8h;
typedef __attribute__((ext_vector_type(4)))  _Float16 v4h;
typedef __attribute__((ext_vector_type(8)))  float    v8f;

// ---- problem constants (match reference) ----
#define CU 50000
#define CI 20000
#define CD 64
#define CS 20
#define CB 8192
#define CN (CU + CI)          // 70000
#define BS (CB * CS)          // 163840 token rows
#define EPS_LN 1e-5f

// padded LDS row pitch in halves (80B: 16B-aligned chunks, conflict-free row stride)
#define APITCH 40
// padded f32 C-stage pitch (rows 8 apart land on different banks: 8*68 % 64 = 32)
#define CPITCH 68

static __device__ __forceinline__ float sigmoidf_(float v) {
  return 1.f / (1.f + __expf(-v));
}

// ---------------------------------------------------------------------------
// e0 = concat(user_emb, item_emb)   [N,64]
// ---------------------------------------------------------------------------
__global__ void k_concat_e0(const float* __restrict__ ue,
                            const float* __restrict__ ie,
                            float* __restrict__ e0) {
  long long idx = (long long)blockIdx.x * blockDim.x + threadIdx.x;
  if (idx >= (long long)CN * CD) return;
  int r = (int)(idx >> 6), d = (int)(idx & 63);
  e0[idx] = (r < CU) ? ue[(size_t)r * CD + d] : ie[(size_t)(r - CU) * CD + d];
}

__global__ void k_zero(float* __restrict__ p, long long n) {
  long long idx = (long long)blockIdx.x * blockDim.x + threadIdx.x;
  if (idx < n) p[idx] = 0.f;
}

// dst[rows[e]] += vals[e] * src[cols[e]]   (one thread per edge*feature)
__global__ void k_scatter(const float* __restrict__ src, float* __restrict__ dst,
                          const int* __restrict__ rows, const int* __restrict__ cols,
                          const float* __restrict__ vals, int E) {
  long long idx = (long long)blockIdx.x * blockDim.x + threadIdx.x;
  if (idx >= (long long)E * CD) return;
  int e = (int)(idx >> 6), d = (int)(idx & 63);
  float contrib = vals[e] * src[(size_t)cols[e] * CD + d];
  __hip_atomic_fetch_add(&dst[(size_t)rows[e] * CD + d], contrib,
                         __ATOMIC_RELAXED, __HIP_MEMORY_SCOPE_AGENT);
}

// mean0 = (e0+e1+e2)/3 ; mean1 = (e0+e2)/2
__global__ void k_means(const float* __restrict__ e0, const float* __restrict__ e1,
                        const float* __restrict__ e2,
                        float* __restrict__ m0, float* __restrict__ m1) {
  long long idx = (long long)blockIdx.x * blockDim.x + threadIdx.x;
  if (idx >= (long long)CN * CD) return;
  float a = e0[idx], b = e1[idx], c = e2[idx];
  m0[idx] = (a + b + c) * (1.f / 3.f);
  m1[idx] = (a + c) * 0.5f;
}

// ---------------------------------------------------------------------------
// token gather: out[b,s,:] = (s==0) ? mean[users[b]] : mean[U + user_item[b,s]]
// ---------------------------------------------------------------------------
__global__ void k_gather_tokens(const float* __restrict__ mean,
                                const int* __restrict__ users,
                                const int* __restrict__ ui,
                                float* __restrict__ out) {
  long long idx = (long long)blockIdx.x * blockDim.x + threadIdx.x;
  if (idx >= (long long)BS * CD) return;
  int b = (int)(idx / (CS * CD));
  int rem = (int)(idx - (long long)b * (CS * CD));
  int s = rem >> 6, d = rem & 63;
  int row = (s == 0) ? users[b] : (CU + ui[b * CS + s]);
  out[idx] = mean[(size_t)row * CD + d];
}

// x[b,s,:] = (s==0) ? user_exp[users[b]] : itemx[user_item[b,s]]
__global__ void k_build_x(const float* __restrict__ itemx,
                          const float* __restrict__ uexp,
                          const int* __restrict__ users,
                          const int* __restrict__ ui,
                          float* __restrict__ out) {
  long long idx = (long long)blockIdx.x * blockDim.x + threadIdx.x;
  if (idx >= (long long)BS * CD) return;
  int b = (int)(idx / (CS * CD));
  int rem = (int)(idx - (long long)b * (CS * CD));
  int s = rem >> 6, d = rem & 63;
  out[idx] = (s == 0) ? uexp[(size_t)users[b] * CD + d]
                      : itemx[(size_t)ui[b * CS + s] * CD + d];
}

__global__ void k_add(const float* __restrict__ a, const float* __restrict__ b,
                      float* __restrict__ o, long long n) {
  long long idx = (long long)blockIdx.x * blockDim.x + threadIdx.x;
  if (idx < n) o[idx] = a[idx] + b[idx];
}

// ---------------------------------------------------------------------------
// Generic WMMA GEMM: C[M,64] = act((A[M,K(lda)] @ W[K,64] + bias) * scale)
// 256 threads = 8 waves; wave w computes rows [blk*128 + 16w, +16), all 64 cols.
// K stepped by 32 through an f16 LDS stage.
//   A tile   : row-major [128][APITCH]   -> A-frag = 2x ds_load_b128 per lane
//   B tile   : transposed [64 n][APITCH] -> B-frag = 2x ds_load_b128 per lane
// Epilogue staged through LDS for coalesced float4 stores.
// act: 0 = none, 1 = sigmoid, 2 = leaky_relu(0.01)
// ---------------------------------------------------------------------------
__global__ __launch_bounds__(256) void k_gemm(
    const float* __restrict__ A, int lda,
    const float* __restrict__ W,      // [K,64] row-major
    const float* __restrict__ bias,   // [64]
    float* __restrict__ C, int ldc,
    int M, int K, float scale, int act) {
  __shared__ __align__(16) _Float16 lA[128 * APITCH];
  __shared__ __align__(16) _Float16 lBt[64 * APITCH];
  __shared__ __align__(16) float    lC[128 * CPITCH];

  const int tid  = threadIdx.x;
  const int lane = tid & 31;
  const int wave = tid >> 5;
  const int hi   = (lane >> 4) & 1;
  const int lo   = lane & 15;
  const int mBase = blockIdx.x * 128;
  const bool fullTile = (mBase + 128 <= M);

  v8f acc[4] = {v8f{}, v8f{}, v8f{}, v8f{}};

  for (int kb = 0; kb < K; kb += 32) {
    // stage A tile 128x32: coalesced float4 loads -> packed v4h (8B) DS stores
    if (fullTile) {
#pragma unroll
      for (int ii = 0; ii < 4; ++ii) {
        int i = tid + ii * 256;
        int r = i >> 3, c4 = i & 7;
        float4 f = *(const float4*)(A + (size_t)(mBase + r) * lda + kb + c4 * 4);
        v4h h = { (_Float16)f.x, (_Float16)f.y, (_Float16)f.z, (_Float16)f.w };
        *(v4h*)(lA + r * APITCH + c4 * 4) = h;
      }
    } else {
      for (int i = tid; i < 128 * 8; i += 256) {
        int r = i >> 3, c4 = i & 7;
        int gr = mBase + r;
        float4 f = (gr < M) ? *(const float4*)(A + (size_t)gr * lda + kb + c4 * 4)
                            : make_float4(0.f, 0.f, 0.f, 0.f);
        v4h h = { (_Float16)f.x, (_Float16)f.y, (_Float16)f.z, (_Float16)f.w };
        *(v4h*)(lA + r * APITCH + c4 * 4) = h;
      }
    }
    // stage B tile 32x64 transposed: lBt[n][k]
#pragma unroll
    for (int ii = 0; ii < 2; ++ii) {
      int i = tid + ii * 256;
      int c = i & 63, g = i >> 6;          // column n, k-group of 4
      const float* wp = W + (size_t)(kb + g * 4) * 64 + c;
      v4h h = { (_Float16)wp[0], (_Float16)wp[64], (_Float16)wp[128], (_Float16)wp[192] };
      *(v4h*)(lBt + c * APITCH + g * 4) = h;
    }
    // prefetch next K tile (global_prefetch_b8)
    if (kb + 32 < K) {
      int gr = mBase + (tid >> 1);
      if (gr < M) __builtin_prefetch(&A[(size_t)gr * lda + kb + 32], 0, 0);
      if (tid < 32) __builtin_prefetch(&W[(size_t)(kb + 32 + tid) * 64], 0, 0);
    }
    __syncthreads();

    // A fragment: two aligned 16B chunks of this lane's row
    const int rowA = (wave << 4) | lo;
    const v8h* arow = (const v8h*)(lA + rowA * APITCH);   // 5 v8h units per row
    v8h a0 = arow[hi];
    v8h a1 = arow[2 + hi];
    v16h af = __builtin_shufflevector(a0, a1, 0, 1, 2, 3, 4, 5, 6, 7,
                                      8, 9, 10, 11, 12, 13, 14, 15);

#pragma unroll
    for (int t = 0; t < 4; ++t) {
      const int n = (t << 4) | lo;
      const v8h* brow = (const v8h*)(lBt + n * APITCH);
      v8h b0 = brow[hi];
      v8h b1 = brow[2 + hi];
      v16h bf = __builtin_shufflevector(b0, b1, 0, 1, 2, 3, 4, 5, 6, 7,
                                        8, 9, 10, 11, 12, 13, 14, 15);
      acc[t] = __builtin_amdgcn_wmma_f32_16x16x32_f16(
          false, af, false, bf, (short)0, acc[t], false, false);
    }
    __syncthreads();
  }

  // epilogue part 1: bias/scale/act applied lane-side, staged into LDS
  // acc VGPR r -> M = r + 8*hi ; N = lo (within 16x16 tile)
#pragma unroll
  for (int t = 0; t < 4; ++t) {
    const int col = (t << 4) | lo;
    const float bv = bias ? bias[col] : 0.f;
    float* dst = lC + ((wave << 4) + (hi ? 8 : 0)) * CPITCH + col;
#pragma unroll
    for (int r = 0; r < 8; ++r) {
      float val = (acc[t][r] + bv) * scale;
      if (act == 1)      val = sigmoidf_(val);
      else if (act == 2) val = val > 0.f ? val : 0.01f * val;
      dst[r * CPITCH] = val;
    }
  }
  __syncthreads();

  // epilogue part 2: coalesced float4 streaming stores
#pragma unroll
  for (int ii = 0; ii < 8; ++ii) {
    int i = tid + ii * 256;                 // 128*16 float4 chunks
    int r = i >> 4, c4 = i & 15;
    int row = mBase + r;
    if (row < M) {
      *(float4*)(C + (size_t)row * ldc + c4 * 4) =
          *(const float4*)(lC + r * CPITCH + c4 * 4);
    }
  }
}

// ---------------------------------------------------------------------------
// attention: per batch b: s = q k^T (q pre-scaled), mask -> -inf, softmax, o = a v
// one wave per batch; S=20, D=64 staged in LDS
// ---------------------------------------------------------------------------
__global__ void k_attn(const float* __restrict__ q, const float* __restrict__ k,
                       const float* __restrict__ v, const unsigned char* __restrict__ mask,
                       float* __restrict__ o) {
  __shared__ float sq[CS * CD], sk[CS * CD], sv[CS * CD];
  const int b = blockIdx.x;
  const int lane = threadIdx.x;
  const size_t base = (size_t)b * CS * CD;
  for (int i = lane; i < CS * CD; i += 32) {
    sq[i] = q[base + i];
    sk[i] = k[base + i];
    sv[i] = v[base + i];
  }
  __syncthreads();
  if (lane < CS) {
    float sc[CS];
    float mx = -INFINITY;
#pragma unroll 1
    for (int kp = 0; kp < CS; ++kp) {
      float s;
      if (mask[b * CS + kp]) {
        s = -INFINITY;
      } else {
        s = 0.f;
        for (int d = 0; d < CD; ++d) s += sq[lane * CD + d] * sk[kp * CD + d];
      }
      sc[kp] = s;
      mx = fmaxf(mx, s);
    }
    float denom = 0.f;
#pragma unroll
    for (int kp = 0; kp < CS; ++kp) {
      float e = expf(sc[kp] - mx);
      sc[kp] = e;
      denom += e;
    }
    float inv = 1.f / denom;
    for (int d = 0; d < CD; ++d) {
      float accv = 0.f;
#pragma unroll
      for (int kp = 0; kp < CS; ++kp) accv += sc[kp] * sv[kp * CD + d];
      o[base + lane * CD + d] = accv * inv;
    }
  }
}

// ---------------------------------------------------------------------------
// layernorm over last dim (64) in-place; 8 rows per 256-thread block
// ---------------------------------------------------------------------------
__global__ void k_ln(float* __restrict__ x, const float* __restrict__ g,
                     const float* __restrict__ b, int rows) {
  int row  = blockIdx.x * 8 + (threadIdx.x >> 5);
  int lane = threadIdx.x & 31;
  if (row >= rows) return;
  float* p = x + (size_t)row * CD;
  float a0 = p[lane], a1 = p[lane + 32];
  float s = a0 + a1;
  for (int m = 16; m; m >>= 1) s += __shfl_xor(s, m, 32);
  float mu = s * (1.f / CD);
  float d0 = a0 - mu, d1 = a1 - mu;
  float vv = d0 * d0 + d1 * d1;
  for (int m = 16; m; m >>= 1) vv += __shfl_xor(vv, m, 32);
  float rs = rsqrtf(vv * (1.f / CD) + EPS_LN);
  p[lane]      = g[lane]      * d0 * rs + b[lane];
  p[lane + 32] = g[lane + 32] * d1 * rs + b[lane + 32];
}

// ---------------------------------------------------------------------------
extern "C" void kernel_launch(void* const* d_in, const int* in_sizes, int n_in,
                              void* d_out, int out_size, void* d_ws, size_t ws_size,
                              hipStream_t stream) {
  (void)n_in; (void)out_size; (void)ws_size;

  const int*   users = (const int*)d_in[0];
  const int*   ui    = (const int*)d_in[1];
  const unsigned char* mask = (const unsigned char*)d_in[2];
  const int*   grows = (const int*)d_in[3];
  const int*   gcols = (const int*)d_in[4];
  const float* gvals = (const float*)d_in[5];
  const float* uemb  = (const float*)d_in[6];
  const float* iemb  = (const float*)d_in[7];
  const float* uexp  = (const float*)d_in[8];
  const int E = in_sizes[3];

  // per-branch input indices:           v    t
  const int IDX_FEAT[2]  = { 9, 10 };
  const int KDIM[2]      = { 4096, 1024 };
  const int IDX_MLPW[2]  = { 11, 27 };
  const int IDX_MLPB[2]  = { 12, 28 };
  const int IDX_LINW[2]  = { 13, 29 };
  const int IDX_LINB[2]  = { 14, 30 };
  const int IDX_DENW[2]  = { 15, 31 };
  const int IDX_DENB[2]  = { 16, 32 };
  const int IDX_QW[2]    = { 17, 33 };
  const int IDX_QB[2]    = { 18, 34 };
  const int IDX_KW[2]    = { 19, 35 };
  const int IDX_KB[2]    = { 20, 36 };
  const int IDX_VW[2]    = { 21, 37 };
  const int IDX_VB[2]    = { 22, 38 };
  const int IDX_OW[2]    = { 23, 39 };
  const int IDX_OB[2]    = { 24, 40 };
  const int IDX_LNG[2]   = { 25, 41 };
  const int IDX_LNB[2]   = { 26, 42 };

  // workspace layout (floats)
  float* wsf = (float*)d_ws;
  size_t off = 0;
  auto alloc = [&](size_t n) { float* p = wsf + off; off += n; return p; };
  const size_t ND = (size_t)CN * CD;
  float* e0    = alloc(ND);
  float* e1    = alloc(ND);
  float* e2    = alloc(ND);
  float* mean0 = alloc(ND);
  float* mean1 = alloc(ND);
  float* itemx = alloc((size_t)CI * CD);
  const size_t TOK = (size_t)BS * CD;
  float* xb   = alloc(TOK);
  float* inp  = alloc(TOK);
  float* src0 = alloc(TOK);
  float* src1 = alloc(TOK);
  float* qb   = alloc(TOK);
  float* kb   = alloc(TOK);
  float* vb   = alloc(TOK);
  float* ob   = alloc(TOK);

  float* out = (float*)d_out;

  auto cdiv = [](long long a, long long b) { return (int)((a + b - 1) / b); };
  const float QSCALE = 0.00125f;  // (64^-0.5)/100

  auto gemm = [&](const float* A, int lda, const float* W, const float* bias,
                  float* C, int ldc, int M, int K, float scale, int act) {
    k_gemm<<<cdiv(M, 128), 256, 0, stream>>>(A, lda, W, bias, C, ldc, M, K, scale, act);
  };

  // ---- LightGCN ----
  k_concat_e0<<<cdiv(ND, 256), 256, 0, stream>>>(uemb, iemb, e0);
  k_zero<<<cdiv(ND, 256), 256, 0, stream>>>(e1, (long long)ND);
  k_scatter<<<cdiv((long long)E * CD, 256), 256, 0, stream>>>(e0, e1, grows, gcols, gvals, E);
  k_zero<<<cdiv(ND, 256), 256, 0, stream>>>(e2, (long long)ND);
  k_scatter<<<cdiv((long long)E * CD, 256), 256, 0, stream>>>(e1, e2, grows, gcols, gvals, E);
  k_means<<<cdiv(ND, 256), 256, 0, stream>>>(e0, e1, e2, mean0, mean1);

  // ---- branches ----
  for (int br = 0; br < 2; ++br) {
    const float* feat   = (const float*)d_in[IDX_FEAT[br]];
    const int    K      = KDIM[br];
    const float* mlp_w  = (const float*)d_in[IDX_MLPW[br]];
    const float* mlp_b  = (const float*)d_in[IDX_MLPB[br]];
    const float* lin_w  = (const float*)d_in[IDX_LINW[br]];
    const float* lin_b  = (const float*)d_in[IDX_LINB[br]];
    const float* den_w  = (const float*)d_in[IDX_DENW[br]];
    const float* den_b  = (const float*)d_in[IDX_DENB[br]];
    const float* q_w    = (const float*)d_in[IDX_QW[br]];
    const float* q_b    = (const float*)d_in[IDX_QB[br]];
    const float* k_w    = (const float*)d_in[IDX_KW[br]];
    const float* k_b    = (const float*)d_in[IDX_KB[br]];
    const float* v_w    = (const float*)d_in[IDX_VW[br]];
    const float* v_b    = (const float*)d_in[IDX_VB[br]];
    const float* o_w    = (const float*)d_in[IDX_OW[br]];
    const float* o_b    = (const float*)d_in[IDX_OB[br]];
    const float* ln_g   = (const float*)d_in[IDX_LNG[br]];
    const float* ln_b   = (const float*)d_in[IDX_LNB[br]];

    // item features -> D (the big GEMM of this branch)
    gemm(feat, K, lin_w, lin_b, itemx, CD, CI, K, 1.f, 0);
    k_build_x<<<cdiv(TOK, 256), 256, 0, stream>>>(itemx, uexp, users, ui, xb);

    // srcs[i] = sigmoid(gather(mean_i) @ mlp + b)
    k_gather_tokens<<<cdiv(TOK, 256), 256, 0, stream>>>(mean0, users, ui, inp);
    gemm(inp, CD, mlp_w, mlp_b, src0, CD, BS, CD, 1.f, 1);
    k_gather_tokens<<<cdiv(TOK, 256), 256, 0, stream>>>(mean1, users, ui, inp);
    gemm(inp, CD, mlp_w, mlp_b, src1, CD, BS, CD, 1.f, 1);

    // encoder: 2 layers
    for (int i = 0; i < 2; ++i) {
      const float* src = (i == 0) ? src0 : src1;
      k_add<<<cdiv(TOK, 256), 256, 0, stream>>>(xb, src, inp, (long long)TOK);
      gemm(inp, CD, q_w + (size_t)i * CD * CD, q_b + (size_t)i * CD, qb, CD, BS, CD, QSCALE, 0);
      gemm(inp, CD, k_w + (size_t)i * CD * CD, k_b + (size_t)i * CD, kb, CD, BS, CD, 1.f, 0);
      gemm(xb,  CD, v_w + (size_t)i * CD * CD, v_b + (size_t)i * CD, vb, CD, BS, CD, 1.f, 0);
      k_attn<<<CB, 32, 0, stream>>>(qb, kb, vb, mask, ob);
      gemm(ob, CD, o_w + (size_t)i * CD * CD, o_b + (size_t)i * CD, xb, CD, BS, CD, 1.f, 0);
      k_ln<<<cdiv(BS, 8), 256, 0, stream>>>(xb, ln_g + (size_t)i * CD, ln_b + (size_t)i * CD, BS);
    }

    // CLS rows (stride S*64) -> dense -> leaky_relu -> out[br]
    gemm(xb, CS * CD, den_w, den_b, out + (size_t)br * CB * CD, CD, CB, CD, 1.f, 2);
  }
}
